// Spikformer_22900765622599
// MI455X (gfx1250) — matrix operand order repaired and checked by reading, not compiled, not run on hardware
//
#include <hip/hip_runtime.h>
#include <hip/hip_bf16.h>

// ---- problem constants ----
#define BB    8
#define NN    1024           // H*W = 32*32
#define CC    256
#define CC2   512
#define HEADS 8
#define HD    32
#define HH    32
#define WW    32

typedef __attribute__((ext_vector_type(2)))  float    v2f;
typedef __attribute__((ext_vector_type(8)))  float    v8f;
typedef __attribute__((ext_vector_type(8)))  _Float16 v8h;
typedef __attribute__((ext_vector_type(16))) _Float16 v16h;
typedef __attribute__((ext_vector_type(4)))  unsigned int v4u;
typedef __attribute__((ext_vector_type(8)))  int      v8i;
typedef __attribute__((ext_vector_type(4)))  int      v4i;

#if defined(__has_builtin)
#if __has_builtin(__builtin_amdgcn_wmma_f32_16x16x4_f32)
#define HAVE_WMMA_F32K4 1
#endif
#if __has_builtin(__builtin_amdgcn_tensor_load_to_lds)
#define HAVE_TDM 1
#endif
#endif

__device__ __forceinline__ v8f wmma_f32_k4(v2f a, v2f b, v8f c) {
#ifdef HAVE_WMMA_F32K4
  return __builtin_amdgcn_wmma_f32_16x16x4_f32(false, a, false, b, (short)0, c,
                                               false, false);
#else
  c[0] += a.x * b.x + a.y * b.y;
  return c;
#endif
}

// ============================================================================
// Kernel A: qk = x @ Wqk + bqk, then *gamma[n] + beta[n].  M=8192 K=256 N=512.
// Block = 8 waves sharing one 16-col B panel (256x16 f32 = 16 KB) staged in
// LDS by the Tensor Data Mover; each wave does a 16x16 tile with a K-loop of
// 64 x V_WMMA_F32_16X16X4_F32, B operand fed from LDS.
// ============================================================================
__global__ __launch_bounds__(256) void k_qk_gemm(
    const float* __restrict__ x, const float* __restrict__ Wqk,
    const float* __restrict__ bqk, const float* __restrict__ gamma,
    const float* __restrict__ beta, float* __restrict__ qk_ws) {
  __shared__ float ldsB[CC * 16];        // 16 KB: B panel, row k x col 0..15
  const int w = threadIdx.x >> 5;
  const int tn = blockIdx.x & 31;        // 32 col tiles (shared per block)
  const int tm = (blockIdx.x >> 5) * 8 + w;  // 512 row tiles
  const int lane = threadIdx.x & 31;
  const int lh = lane & 15;
  const int hs = lane >> 4;

#ifdef HAVE_TDM
  if (threadIdx.x < 32) {
    // --- Tensor DMA descriptor (D#): 2D tile 16 x 256 of Wqk (512-wide) ---
    unsigned long long ga =
        (unsigned long long)(const void*)(Wqk + (size_t)tn * 16);
    v4u g0;
    g0[0] = 1u;                                   // count=1, user mode
    g0[1] = 0u;                                   // lds_addr = 0 (ldsB base)
    g0[2] = (unsigned)(ga & 0xffffffffu);         // global_addr[31:0]
    g0[3] = (unsigned)((ga >> 32) & 0x1ffffffu)   // global_addr[56:32]
            | (2u << 30);                         // type = 2 ("image")
    v8i g1;
    g1[0] = (int)(2u << 16);    // wg_mask=0, data_size=2 (4 bytes)
    g1[1] = (int)(512u << 16);  // tensor_dim0 = 512 (low16 in [31:16])
    g1[2] = (int)(256u << 16);  // tensor_dim0 hi=0 | tensor_dim1 = 256
    g1[3] = (int)(16u << 16);   // tensor_dim1 hi=0 | tile_dim0 = 16
    g1[4] = 256;                // tile_dim1 = 256, tile_dim2 = 0
    g1[5] = 512;                // tensor_dim0_stride[31:0] = 512
    g1[6] = 0;                  // stride hi, tensor_dim1_stride lo
    g1[7] = 0;
    v4i g2 = {0, 0, 0, 0}, g3 = {0, 0, 0, 0};
    v8i g4 = {0, 0, 0, 0, 0, 0, 0, 0};
    // 6-arg form (clang-23 / therock-10.0 toolchain)
    __builtin_amdgcn_tensor_load_to_lds(g0, g1, g2, g3, g4, 0);
    __builtin_amdgcn_s_wait_tensorcnt((short)0);
  }
  // The TDM writes ldsB behind the compiler's back (the builtin carries the
  // LDS address as an integer, not a pointer operand), so LLVM sees a
  // never-stored LDS global and folds its loads to undef.  Defeat that with
  // an opaque, never-executed store + a memory-clobber fence.
  {
    unsigned guard;
    asm volatile("s_mov_b32 %0, 0" : "=s"(guard));
    if (guard) ldsB[guard & (CC * 16 - 1)] = 0.0f;
    asm volatile("" ::: "memory");
  }
#else
  for (int i = threadIdx.x; i < CC * 16; i += 256)
    ldsB[i] = Wqk[(size_t)(i >> 4) * CC2 + tn * 16 + (i & 15)];
#endif
  __syncthreads();

  // A layout (16x4 f32): lanes 0-15 rows M hold K=0,1; lanes 16-31 K=2,3.
  const float* Arow = x + (size_t)(tm * 16 + lh) * CC;
  const int colB = tn * 16 + lh;
  v8f acc = {};
#pragma unroll 4
  for (int kk = 0; kk < CC; kk += 4) {
    v2f a = *(const v2f*)(Arow + kk + hs * 2);
    v2f b;
    b.x = ldsB[(kk + hs * 2 + 0) * 16 + lh];
    b.y = ldsB[(kk + hs * 2 + 1) * 16 + lh];
    acc = wmma_f32_k4(a, b, acc);
  }
  const float bq = bqk[colB];
#pragma unroll
  for (int j = 0; j < 8; ++j) {
    const int r = tm * 16 + hs * 8 + j;  // global row in (B*N)
    const int npos = r & (NN - 1);
    float v = acc[j] + bq;
    v = v * gamma[npos] + beta[npos];
    qk_ws[(size_t)r * CC2 + colB] = v;
  }
}

// ============================================================================
// Kernel B: LIF scan over batch axis + elu(spike)+1 + 2D RoPE.
// One thread per (n, channel-pair); pairs 0..127 -> q, 128..255 -> k.
// ============================================================================
__global__ __launch_bounds__(256) void k_lif_rope(
    const float* __restrict__ qk_ws, float* __restrict__ qhf,
    float* __restrict__ khf, _Float16* __restrict__ qh16,
    _Float16* __restrict__ kh16, float* __restrict__ qrope,
    float* __restrict__ krope) {
  const int idx = blockIdx.x * blockDim.x + threadIdx.x;  // < 1024*256
  const int n = idx >> 8;
  const int p = idx & 255;
  const bool isK = p >= 128;
  const int pp = isK ? (p - 128) : p;
  const int ch0 = (isK ? CC : 0) + pp * 2;

  const int hc = n >> 5, wc = n & 31;
  const float j = (float)((pp < 64) ? pp : (pp - 64));
  const float theta = exp2f(-(j * (1.0f / 64.0f)) * 13.28771237954945f);
  const float ang = ((pp < 64) ? (float)hc : (float)wc) * theta;
  float sn, cs;
  __sincosf(ang, &sn, &cs);

  const int h = pp >> 4;
  const int d0 = (pp << 1) & 31;

  float* fdst = isK ? khf : qhf;
  _Float16* hdst = isK ? kh16 : qh16;
  float* rdst = isK ? krope : qrope;

  float v0 = 0.f, v1 = 0.f;
  for (int b = 0; b < BB; ++b) {
    v2f xv = *(const v2f*)(qk_ws + (size_t)(b * NN + n) * CC2 + ch0);
    v0 = 0.5f * (v0 + xv.x);
    v1 = 0.5f * (v1 + xv.y);
    const bool f0 = v0 >= 1.0f, f1 = v1 >= 1.0f;
    const float s0 = f0 ? 2.0f : 1.0f;
    const float s1 = f1 ? 2.0f : 1.0f;
    if (f0) v0 = 0.f;
    if (f1) v1 = 0.f;

    const size_t base = ((((size_t)b * HEADS + h) * NN) + n) * HD + d0;
    fdst[base] = s0;
    fdst[base + 1] = s1;
    hdst[base] = (_Float16)s0;
    hdst[base + 1] = (_Float16)s1;
    rdst[base] = cs * s0 - sn * s1;
    rdst[base + 1] = sn * s0 + cs * s1;
  }
}

// ============================================================================
// Kernel C: atten[b,h] = qh @ kh^T  (1024x1024x32), exact in f16.
// One wave owns a 16x64 strip: 1 A load feeds 4 x V_WMMA_F32_16X16X32_F16
// (K=32 = head_dim in one WMMA).  Results repacked through LDS so every
// global store is b128 and each half-wave covers a contiguous 256 B segment —
// this is the 256 MB output that sets the HBM roofline.
// ============================================================================
__global__ __launch_bounds__(256) void k_atten(
    const _Float16* __restrict__ qh16, const _Float16* __restrict__ kh16,
    float* __restrict__ atten) {
  __shared__ float lds[8][16][64];       // 32 KB
  const int w = threadIdx.x >> 5;
  const int wave = blockIdx.x * 8 + w;
  const int bh = wave >> 10;             // 1024 strips per (b,h)
  const int s = wave & 1023;
  const int tm = s >> 4;                 // 64 row tiles
  const int ts = s & 15;                 // 16 col strips of 64
  const int lane = threadIdx.x & 31;
  const int lh = lane & 15;
  const int hs = lane >> 4;

  const _Float16* Q = qh16 + (size_t)bh * NN * HD;
  const _Float16* K = kh16 + (size_t)bh * NN * HD;

  // A (16x32 f16): lane<16 row M holds K0..15; lane>=16 K16..31
  const _Float16* qrow = Q + (size_t)(tm * 16 + lh) * HD + hs * 16;
  v8h alo = *(const v8h*)(qrow);
  v8h ahi = *(const v8h*)(qrow + 8);
  v16h a;
#pragma unroll
  for (int i = 0; i < 8; ++i) { a[i] = alo[i]; a[8 + i] = ahi[i]; }

  v8f acc[4];
#pragma unroll
  for (int t = 0; t < 4; ++t) {
    const int col = ts * 64 + t * 16 + lh;
    v16h b = *(const v16h*)(K + (size_t)col * HD + hs * 16);
    v8f c = {};
    acc[t] = __builtin_amdgcn_wmma_f32_16x16x32_f16(false, a, false, b,
                                                    (short)0, c, false, false);
  }

  // D layout -> LDS strip [row 0..15][col 0..63]
#pragma unroll
  for (int t = 0; t < 4; ++t)
#pragma unroll
    for (int j = 0; j < 8; ++j)
      lds[w][hs * 8 + j][t * 16 + lh] = acc[t][j];

  float* A = atten + (size_t)bh * NN * NN + (size_t)(tm * 16) * NN + ts * 64;
#pragma unroll
  for (int k = 0; k < 8; ++k) {
    const int row = k * 2 + hs;          // half-wave = one full 256 B row seg
    const int ch = lh;                   // 16 chunks of 16 B per row
    float4 vchunk = *(const float4*)&lds[w][row][ch * 4];
    *(float4*)(A + (size_t)row * NN + ch * 4) = vchunk;
  }
}

// ============================================================================
// Kernel D: kmean over n, then z[n] = 1/(qh[n]·kmean + 1e-6).  1 block/(b,h).
// ============================================================================
__global__ __launch_bounds__(256) void k_z(const float* __restrict__ qhf,
                                           const float* __restrict__ khf,
                                           float* __restrict__ zbuf) {
  __shared__ float part[256];
  __shared__ float kmean[HD];
  const int bh = blockIdx.x;
  const int t = threadIdx.x;
  const int d = t & 31, chunk = t >> 5;
  const float* K = khf + (size_t)bh * NN * HD;
  const float* Q = qhf + (size_t)bh * NN * HD;

  float s = 0.f;
  for (int n = chunk * 128; n < chunk * 128 + 128; ++n)
    s += K[(size_t)n * HD + d];
  part[t] = s;
  __syncthreads();
  if (t < HD) {
    float tot = 0.f;
#pragma unroll
    for (int c = 0; c < 8; ++c) tot += part[c * 32 + t];
    kmean[t] = tot * (1.0f / (float)NN);
  }
  __syncthreads();
  for (int n = t; n < NN; n += 256) {
    float dot = 0.f;
#pragma unroll
    for (int d2 = 0; d2 < HD; ++d2) dot += Q[(size_t)n * HD + d2] * kmean[d2];
    zbuf[(size_t)bh * NN + n] = 1.0f / (dot + 1e-6f);
  }
}

// ============================================================================
// Kernel E: kv[b,h] = k_rope^T @ vh / N   (32x32, K=1024).  1 block/(b,h).
// ============================================================================
__global__ __launch_bounds__(1024) void k_kv(const float* __restrict__ krope,
                                             const float* __restrict__ x,
                                             float* __restrict__ kv) {
  const int bh = blockIdx.x;
  const int d = threadIdx.x >> 5;
  const int e = threadIdx.x & 31;
  const int b = bh >> 3, h = bh & 7;
  const float* KR = krope + (size_t)bh * NN * HD;
  const float* Xb = x + (size_t)b * NN * CC + h * HD;
  float s = 0.f;
  for (int n = 0; n < NN; ++n)
    s += KR[(size_t)n * HD + d] * Xb[(size_t)n * CC + e];
  kv[((size_t)bh * HD + d) * HD + e] = s * (1.0f / (float)NN);
}

// ============================================================================
// Kernel F: out = (q_rope @ kv) * z, written to (b,n,c).  f32 WMMA,
// M=1024 N=32 K=32 per (b,h); K-loop of 8 x V_WMMA_F32_16X16X4_F32.
// ============================================================================
__global__ __launch_bounds__(256) void k_out(const float* __restrict__ qrope,
                                             const float* __restrict__ kv,
                                             const float* __restrict__ zbuf,
                                             float* __restrict__ out) {
  const int wave = blockIdx.x * 8 + (threadIdx.x >> 5);
  const int bh = wave >> 7;
  const int t = wave & 127;
  const int tm = t >> 1, tn = t & 1;
  const int lane = threadIdx.x & 31;
  const int lh = lane & 15;
  const int hs = lane >> 4;
  const int b = bh >> 3, h = bh & 7;

  const float* Arow = qrope + ((size_t)bh * NN + tm * 16 + lh) * HD;
  const float* KV = kv + (size_t)bh * HD * HD;
  const int col = tn * 16 + lh;

  v8f acc = {};
#pragma unroll
  for (int kk = 0; kk < HD; kk += 4) {
    v2f a = *(const v2f*)(Arow + kk + hs * 2);
    v2f bv;
    bv.x = KV[(size_t)(kk + hs * 2 + 0) * HD + col];
    bv.y = KV[(size_t)(kk + hs * 2 + 1) * HD + col];
    acc = wmma_f32_k4(a, bv, acc);
  }
#pragma unroll
  for (int j = 0; j < 8; ++j) {
    const int row = tm * 16 + hs * 8 + j;
    const float z = zbuf[(size_t)bh * NN + row];
    out[((size_t)b * NN + row) * CC + h * HD + col] = acc[j] * z;
  }
}

// ============================================================================
// Kernel G: depthwise 3x3 LePE + bias, accumulated into out.
// Channel is the lane-fastest index: all 9 taps, bias and the out RMW are
// fully coalesced.
// ============================================================================
__global__ __launch_bounds__(256) void k_lepe(const float* __restrict__ x,
                                              const float* __restrict__ lw,
                                              const float* __restrict__ lb,
                                              float* __restrict__ out) {
  const int idx = blockIdx.x * blockDim.x + threadIdx.x;  // < 8*1024*256
  const int c = idx & 255;
  const int pos = (idx >> 8) & 1023;
  const int b = idx >> 18;
  const int hh = pos >> 5, ww = pos & 31;

  float acc = lb[c];
#pragma unroll
  for (int i = 0; i < 3; ++i) {
#pragma unroll
    for (int jx = 0; jx < 3; ++jx) {
      const int y = hh + i - 1, xw = ww + jx - 1;
      if (y >= 0 && y < HH && xw >= 0 && xw < WW)
        acc += lw[c * 9 + i * 3 + jx] *
               x[((size_t)b * NN + y * WW + xw) * CC + c];
    }
  }
  out[((size_t)b * NN + pos) * CC + c] += acc;
}

// ============================================================================
// Kernel H: v_img = transpose of x, LDS-tiled 32x32 so both global sides are
// coalesced.
// ============================================================================
__global__ __launch_bounds__(256) void k_vimg(const float* __restrict__ x,
                                              float* __restrict__ v_img) {
  __shared__ float tile[32][33];
  const int b = blockIdx.x >> 8;          // 8
  const int pt = (blockIdx.x >> 3) & 31;  // 32 pos tiles
  const int ct = blockIdx.x & 7;          // 8 channel tiles
  const int tx = threadIdx.x & 31;
  const int ty = threadIdx.x >> 5;        // 0..7
#pragma unroll
  for (int i = 0; i < 4; ++i) {
    const int pos = pt * 32 + ty + i * 8;
    tile[ty + i * 8][tx] = x[((size_t)b * NN + pos) * CC + ct * 32 + tx];
  }
  __syncthreads();
#pragma unroll
  for (int i = 0; i < 4; ++i) {
    const int c = ct * 32 + ty + i * 8;
    v_img[((size_t)b * CC + c) * NN + pt * 32 + tx] = tile[tx][ty + i * 8];
  }
}

// ============================================================================
extern "C" void kernel_launch(void* const* d_in, const int* in_sizes, int n_in,
                              void* d_out, int out_size, void* d_ws,
                              size_t ws_size, hipStream_t stream) {
  const float* x = (const float*)d_in[0];
  const float* Wqk = (const float*)d_in[1];
  const float* bqk = (const float*)d_in[2];
  const float* gamma = (const float*)d_in[3];
  const float* beta = (const float*)d_in[4];
  const float* lw = (const float*)d_in[5];
  const float* lb = (const float*)d_in[6];

  float* outf = (float*)d_out;
  float* o_out = outf;                         // (8,1024,256)
  float* o_qh = outf + 2097152;                // (8,8,1024,32)
  float* o_kh = outf + 4194304;                // (8,8,1024,32)
  float* o_vimg = outf + 6291456;              // (8,256,32,32)
  float* o_att = outf + 8388608;               // (8,8,1024,1024)

  float* ws = (float*)d_ws;
  float* qk_ws = ws;                           // 4,194,304 f32
  float* qrope = ws + 4194304;                 // 2,097,152 f32
  float* krope = qrope + 2097152;              // 2,097,152 f32
  _Float16* qh16 = (_Float16*)(krope + 2097152);  // 2,097,152 f16
  _Float16* kh16 = qh16 + 2097152;                // 2,097,152 f16
  float* zbuf = (float*)(kh16 + 2097152);      // 65,536 f32
  float* kvbuf = zbuf + 65536;                 // 65,536 f32

  k_qk_gemm<<<2048, 256, 0, stream>>>(x, Wqk, bqk, gamma, beta, qk_ws);
  k_lif_rope<<<1024, 256, 0, stream>>>(qk_ws, o_qh, o_kh, qh16, kh16, qrope,
                                       krope);
  k_atten<<<8192, 256, 0, stream>>>(qh16, kh16, o_att);
  k_z<<<64, 256, 0, stream>>>(o_qh, o_kh, zbuf);
  k_kv<<<64, 1024, 0, stream>>>(krope, x, kvbuf);
  k_out<<<1024, 256, 0, stream>>>(qrope, kvbuf, zbuf, o_out);
  k_lepe<<<8192, 256, 0, stream>>>(x, lw, lb, o_out);
  k_vimg<<<2048, 256, 0, stream>>>(x, o_vimg);
}
